// DecisionMambaActor_85109071938153
// MI455X (gfx1250) — matrix operand order repaired and checked by reading, not compiled, not run on hardware
//
#include <hip/hip_runtime.h>
#include <hip/hip_bf16.h>
#include <cstddef>
#include <cstdint>

typedef _Float16 v16h __attribute__((ext_vector_type(16)));
typedef _Float16 v8h  __attribute__((ext_vector_type(8)));
typedef float    v8f  __attribute__((ext_vector_type(8)));
typedef uint32_t u32x4 __attribute__((ext_vector_type(4)));
typedef int32_t  i32x8 __attribute__((ext_vector_type(8)));
typedef int32_t  i32x4 __attribute__((ext_vector_type(4)));

#define DM    256
#define DIN   512
#define DST   16
#define DCONV 4
#define DTR   16
#define SDIM  128
#define BB    8
#define TT    2048
#define MROWS (BB*TT)           // 16384
#define ADIM  18

#if defined(__has_builtin)
#if __has_builtin(__builtin_amdgcn_tensor_load_to_lds) && __has_builtin(__builtin_amdgcn_s_wait_tensorcnt)
#define HAVE_TDM 1
#endif
#endif

// ---------------------------------------------------------------------------
// small elementwise helpers
// ---------------------------------------------------------------------------
__global__ void cvt_f32_to_f16_kernel(const float* __restrict__ s,
                                      _Float16* __restrict__ d, int n) {
    int i = blockIdx.x * blockDim.x + threadIdx.x;
    if (i < n) d[i] = (_Float16)s[i];
}

__global__ void aneg_kernel(const float* __restrict__ alog,
                            float* __restrict__ aneg, int n) {
    int i = blockIdx.x * blockDim.x + threadIdx.x;
    if (i < n) aneg[i] = -__expf(alog[i]);
}

// ---------------------------------------------------------------------------
// fragment load helpers (layouts per CDNA5 ISA 7.12.2, wave32)
// A 16x32 f16: lane l -> row m=l&15, K-halves [8*(l>>4)..+8) and +16
// B 32x16 (from row-major W[N,K]): lane l -> col n=l&15, K-halves [8*(l>>4)..+8) and +16
// ---------------------------------------------------------------------------
__device__ inline v16h load_a_frag(const _Float16* Arow, int k0, int ka) {
    v8h alo = *(const v8h*)(Arow + k0 + ka);
    v8h ahi = *(const v8h*)(Arow + k0 + ka + 16);
    union { v16h v; v8h h[2]; } ua;
    ua.h[0] = alo; ua.h[1] = ahi;
    return ua.v;
}

// ---------------------------------------------------------------------------
// Generic f16 WMMA GEMM:  C[M,N] = A[M,K] @ W[N,K]^T   (f32 accum)
// block = 256 threads = 8 waves arranged 2 (M) x 4 (N); each wave owns one
// 16x16 C tile; K-loop in steps of 32 feeding v_wmma_f32_16x16x32_f16.
// ---------------------------------------------------------------------------
__global__ void gemm_wmma_f16_kernel(const _Float16* __restrict__ A,
                                     const _Float16* __restrict__ W,
                                     float* __restrict__ Cf,
                                     _Float16* __restrict__ Ch,
                                     int M, int N, int K) {
    const int lane = threadIdx.x & 31;
    const int wave = threadIdx.x >> 5;          // 0..7
    const int wm   = wave >> 2;                 // 0..1
    const int wn   = wave & 3;                  // 0..3
    const int tm   = blockIdx.x * 2 + wm;       // 16-row tile index
    const int tn   = blockIdx.y * 4 + wn;       // 16-col tile index
    if (tm * 16 >= M || tn * 16 >= N) return;   // uniform per wave

    const int m  = tm * 16 + (lane & 15);
    const int n  = tn * 16 + (lane & 15);
    const int ka = (lane >> 4) * 8;
    const int kb = (lane >> 4) * 8;

    v8f acc = {};
    const _Float16* Arow = A + (size_t)m * K;
    const _Float16* Wrow = W + (size_t)n * K;
    for (int k0 = 0; k0 < K; k0 += 32) {
        v16h af = load_a_frag(Arow, k0, ka);
        v16h bf = load_a_frag(Wrow, k0, kb);
        acc = __builtin_amdgcn_wmma_f32_16x16x32_f16(
                  false, af, false, bf, (short)0, acc, false, false);
    }

    const int row0 = tm * 16 + (lane >> 4) * 8;
    const int col  = tn * 16 + (lane & 15);
    if (Cf) {
        #pragma unroll
        for (int v = 0; v < 8; ++v)
            Cf[(size_t)(row0 + v) * N + col] = acc[v];
    }
    if (Ch) {
        #pragma unroll
        for (int v = 0; v < 8; ++v)
            Ch[(size_t)(row0 + v) * N + col] = (_Float16)acc[v];
    }
}

// ---------------------------------------------------------------------------
// Specialized in_proj GEMM: C[M,1024] = A[M,256] @ W[1024,256]^T, f16 out.
// Block tile 64 (M) x 128 (N); 8 waves (2x4); 2x2 WMMA tiles per wave so
// each A/B fragment feeds two v_wmma ops. The 128x256 f16 weight slab
// (64 KB) is DMA'd into LDS once per block by the Tensor Data Mover
// (TENSOR_LOAD_TO_LDS, waited with s_wait_tensorcnt), then B fragments are
// served from LDS as two 16B ds loads per fragment (k0+kb and k0+kb+16).
//
// `never` is always 0 at runtime; the guarded store makes the LDS array
// visibly written so the compiler cannot fold the ds loads to poison
// (the TDM intrinsic carries no pointer, so AA would otherwise conclude
// sW is never stored).
// ---------------------------------------------------------------------------
#define IP_K   256
#define IP_NT  128            // N rows staged in LDS per block

__global__ void gemm_inproj_wmma_tdm_kernel(const _Float16* __restrict__ A,
                                            const _Float16* __restrict__ W,
                                            _Float16* __restrict__ Ch,
                                            int M, int N, int never) {
    __shared__ _Float16 sW[IP_NT * IP_K];       // 64 KB, LDS offset 0

    const int lane = threadIdx.x & 31;
    const int wave = threadIdx.x >> 5;          // 0..7
    const int wm   = wave >> 2;                 // 0..1 (32-row sub-tile)
    const int wn   = wave & 3;                  // 0..3 (32-col sub-tile)
    const int mbase = blockIdx.x * 64;
    const int nbase = blockIdx.y * IP_NT;

    if (never) sW[threadIdx.x] = (_Float16)(float)never;   // opaque store

    // ---- stage W[nbase .. nbase+127][0..255] into LDS ----
#ifdef HAVE_TDM
    if (threadIdx.x < 32) {
        const uint64_t ga = (uint64_t)(uintptr_t)(W + (size_t)nbase * IP_K);
        u32x4 g0;
        g0[0] = 1u;                                   // count=1, no gather
        g0[1] = 0u;                                   // lds_addr = 0 (sW)
        g0[2] = (uint32_t)ga;                         // global_addr lo
        g0[3] = (uint32_t)((ga >> 32) & 0x01FFFFFFu) | 0x80000000u; // type=2
        i32x8 g1;
        g1[0] = 0x00010000;                           // data_size=1 (2B)
        g1[1] = (int32_t)((uint32_t)IP_K << 16);      // tensor_dim0 = 256
        g1[2] = (int32_t)((uint32_t)N << 16);         // tensor_dim1 = 1024
        g1[3] = (int32_t)((uint32_t)IP_K << 16);      // tile_dim0 = 256
        g1[4] = IP_NT;                                // tile_dim1 = 128
        g1[5] = IP_K;                                 // dim0_stride = 256
        g1[6] = 0;
        g1[7] = 0;
        i32x4 g2 = {};                                // <=2D: groups 2/3 unused
        i32x4 g3 = {};
        i32x8 g4 = {};                                // trailing group (unused)
        __builtin_amdgcn_tensor_load_to_lds(g0, g1, g2, g3, g4, 0);
        __builtin_amdgcn_s_wait_tensorcnt(0);
    }
    __syncthreads();
#else
    // fallback: cooperative copy (4B per thread per iter)
    {
        const uint32_t* src = (const uint32_t*)(W + (size_t)nbase * IP_K);
        uint32_t* dst = (uint32_t*)sW;
        const int words = IP_NT * IP_K / 2;           // 16384 u32
        for (int i = threadIdx.x; i < words; i += 256) dst[i] = src[i];
        __syncthreads();
    }
#endif

    const int ka = (lane >> 4) * 8;
    const int kb = (lane >> 4) * 8;
    const _Float16* Ar0 = A + (size_t)(mbase + wm * 32 +      (lane & 15)) * IP_K;
    const _Float16* Ar1 = A + (size_t)(mbase + wm * 32 + 16 + (lane & 15)) * IP_K;
    const _Float16* Bl0 = sW + (size_t)(wn * 32 +      (lane & 15)) * IP_K;
    const _Float16* Bl1 = sW + (size_t)(wn * 32 + 16 + (lane & 15)) * IP_K;

    v8f acc00 = {}, acc01 = {}, acc10 = {}, acc11 = {};
    #pragma unroll
    for (int k0 = 0; k0 < IP_K; k0 += 32) {
        v16h a0 = load_a_frag(Ar0, k0, ka);
        v16h a1 = load_a_frag(Ar1, k0, ka);
        v16h b0 = load_a_frag(Bl0, k0, kb);      // ds_load_b128 x2
        v16h b1 = load_a_frag(Bl1, k0, kb);
        acc00 = __builtin_amdgcn_wmma_f32_16x16x32_f16(false, a0, false, b0, (short)0, acc00, false, false);
        acc01 = __builtin_amdgcn_wmma_f32_16x16x32_f16(false, a0, false, b1, (short)0, acc01, false, false);
        acc10 = __builtin_amdgcn_wmma_f32_16x16x32_f16(false, a1, false, b0, (short)0, acc10, false, false);
        acc11 = __builtin_amdgcn_wmma_f32_16x16x32_f16(false, a1, false, b1, (short)0, acc11, false, false);
    }

    const int r8  = (lane >> 4) * 8;
    const int cl  = lane & 15;
    v8f accs[4] = {acc00, acc01, acc10, acc11};
    #pragma unroll
    for (int q = 0; q < 4; ++q) {
        const int row0 = mbase + wm * 32 + (q >> 1) * 16 + r8;
        const int col  = nbase + wn * 32 + (q & 1) * 16 + cl;
        #pragma unroll
        for (int v = 0; v < 8; ++v)
            Ch[(size_t)(row0 + v) * N + col] = (_Float16)accs[q][v];
    }
}

// ---------------------------------------------------------------------------
// embed epilogue: tokens_h = f16((stateGEMM + bs + br + rtg*Wr + pos) * mask)
// ---------------------------------------------------------------------------
__global__ void embed_epilogue_kernel(const float* __restrict__ tp,
                                      const float* __restrict__ rtg,
                                      const float* __restrict__ mask,
                                      const float* __restrict__ Wr,
                                      const float* __restrict__ bs,
                                      const float* __restrict__ br,
                                      const float* __restrict__ pos,
                                      _Float16* __restrict__ th, int total) {
    int i = blockIdx.x * blockDim.x + threadIdx.x;
    if (i >= total) return;
    int m  = i & (DM - 1);
    int bt = i >> 8;
    int t  = bt & (TT - 1);
    float v = tp[i] + bs[m] + br[m] + rtg[bt] * Wr[m] + pos[t * DM + m];
    th[i] = (_Float16)(v * mask[bt]);
}

// ---------------------------------------------------------------------------
// causal depthwise conv1d + bias + SiLU on x half of xz  -> xconv (f16)
// ---------------------------------------------------------------------------
__global__ void conv_silu_kernel(const _Float16* __restrict__ xz,
                                 const float* __restrict__ cw,
                                 const float* __restrict__ cb,
                                 _Float16* __restrict__ xo, int total) {
    int i = blockIdx.x * blockDim.x + threadIdx.x;
    if (i >= total) return;
    int d  = i & (DIN - 1);
    int bt = i >> 9;
    int t  = bt & (TT - 1);
    int b  = bt >> 11;
    float acc = cb[d];
    #pragma unroll
    for (int j = 0; j < DCONV; ++j) {
        int tt = t - (DCONV - 1) + j;
        if (tt >= 0)
            acc += (float)xz[((size_t)(b * TT + tt)) * (2 * DIN) + d] * cw[d * DCONV + j];
    }
    float sig = 1.f / (1.f + __expf(-acc));
    xo[i] = (_Float16)(acc * sig);
}

// ---------------------------------------------------------------------------
// dt = softplus(dbc[:, :16] @ dt_proj_w.T + dt_proj_b)   (K=16 -> VALU)
// ---------------------------------------------------------------------------
__global__ void dt_kernel(const float* __restrict__ dbc,
                          const float* __restrict__ dw,
                          const float* __restrict__ db,
                          float* __restrict__ dt, int total) {
    int i = blockIdx.x * blockDim.x + threadIdx.x;
    if (i >= total) return;
    int d  = i & (DIN - 1);
    int bt = i >> 9;
    float acc = db[d];
    const float* row = dbc + (size_t)bt * (DTR + 2 * DST);
    #pragma unroll
    for (int r = 0; r < DTR; ++r) acc += row[r] * dw[d * DTR + r];
    dt[i] = (acc > 20.f) ? acc : log1pf(__expf(acc));
}

// ---------------------------------------------------------------------------
// last valid index per batch from mask
// ---------------------------------------------------------------------------
__global__ void lastidx_kernel(const float* __restrict__ mask,
                               int* __restrict__ idx) {
    __shared__ float s[256];
    int b = blockIdx.x;
    float acc = 0.f;
    for (int t = threadIdx.x; t < TT; t += 256) acc += mask[b * TT + t];
    s[threadIdx.x] = acc;
    __syncthreads();
    for (int st = 128; st > 0; st >>= 1) {
        if (threadIdx.x < st) s[threadIdx.x] += s[threadIdx.x + st];
        __syncthreads();
    }
    if (threadIdx.x == 0) {
        float v = s[0];
        v = fminf(fmaxf(v, 1.f), (float)TT);
        idx[b] = (int)v - 1;
    }
}

// ---------------------------------------------------------------------------
// selective scan, fused with dA/dBx computation, D-skip and SiLU(z) gate.
// ---------------------------------------------------------------------------
__global__ void scan_kernel(const float* __restrict__ dt,
                            const _Float16* __restrict__ xconv,
                            const _Float16* __restrict__ xz,
                            const float* __restrict__ dbc,
                            const float* __restrict__ aneg,
                            const float* __restrict__ Dv,
                            _Float16* __restrict__ yg) {
    __shared__ float sBC[2 * DST];
    const int b = blockIdx.x >> 1;
    const int d = ((blockIdx.x & 1) << 8) + threadIdx.x;

    float a[DST], h[DST];
    #pragma unroll
    for (int s = 0; s < DST; ++s) { a[s] = aneg[d * DST + s]; h[s] = 0.f; }
    const float Dd = Dv[d];

    for (int t = 0; t < TT; ++t) {
        const size_t bt = (size_t)b * TT + t;
        if (threadIdx.x < 2 * DST)
            sBC[threadIdx.x] = dbc[bt * (DTR + 2 * DST) + DTR + threadIdx.x];
        __syncthreads();

        const float dtv = dt[bt * DIN + d];
        const float xv  = (float)xconv[bt * DIN + d];
        const float zv  = (float)xz[bt * (2 * DIN) + DIN + d];
        const float dtx = dtv * xv;
        float y = 0.f;
        #pragma unroll
        for (int s = 0; s < DST; ++s) {
            float dA = __expf(dtv * a[s]);
            h[s] = dA * h[s] + dtx * sBC[s];
            y += h[s] * sBC[DST + s];
        }
        const float sig = 1.f / (1.f + __expf(-zv));
        yg[bt * DIN + d] = (_Float16)((y + xv * Dd) * zv * sig);
        __syncthreads();
    }
}

// ---------------------------------------------------------------------------
// final: out_proj (only the 8 selected rows) + LayerNorm + head
// ---------------------------------------------------------------------------
__global__ void head_kernel(const _Float16* __restrict__ yg,
                            const float* __restrict__ out_w,
                            const float* __restrict__ ln_g,
                            const float* __restrict__ ln_b,
                            const float* __restrict__ head_w,
                            const float* __restrict__ head_b,
                            const int* __restrict__ lastidx,
                            float* __restrict__ out) {
    __shared__ float s_h[DM], s_r[DM], s_ln[DM];
    const int b = blockIdx.x;
    const int m = threadIdx.x;
    const int tl = lastidx[b];
    const _Float16* yrow = yg + ((size_t)(b * TT + tl)) * DIN;

    float acc = 0.f;
    const float* wrow = out_w + (size_t)m * DIN;
    for (int dd = 0; dd < DIN; ++dd) acc += (float)yrow[dd] * wrow[dd];
    s_h[m] = acc; s_r[m] = acc;
    __syncthreads();
    for (int st = 128; st > 0; st >>= 1) {
        if (m < st) s_r[m] += s_r[m + st];
        __syncthreads();
    }
    const float mu = s_r[0] / (float)DM;
    __syncthreads();
    float dv = s_h[m] - mu;
    s_r[m] = dv * dv;
    __syncthreads();
    for (int st = 128; st > 0; st >>= 1) {
        if (m < st) s_r[m] += s_r[m + st];
        __syncthreads();
    }
    const float inv = rsqrtf(s_r[0] / (float)DM + 1e-5f);
    s_ln[m] = (s_h[m] - mu) * inv * ln_g[m] + ln_b[m];
    __syncthreads();
    if (m < ADIM) {
        float l = head_b[m];
        for (int j = 0; j < DM; ++j) l += s_ln[j] * head_w[m * DM + j];
        out[b * ADIM + m] = l;
    }
}

// ---------------------------------------------------------------------------
// launch
// ---------------------------------------------------------------------------
extern "C" void kernel_launch(void* const* d_in, const int* in_sizes, int n_in,
                              void* d_out, int out_size, void* d_ws, size_t ws_size,
                              hipStream_t stream) {
    const float* state   = (const float*)d_in[0];
    const float* rtg     = (const float*)d_in[1];
    const float* mask    = (const float*)d_in[2];
    const float* Ws      = (const float*)d_in[3];
    const float* bs      = (const float*)d_in[4];
    const float* Wr      = (const float*)d_in[5];
    const float* br      = (const float*)d_in[6];
    const float* pos     = (const float*)d_in[7];
    const float* inprj   = (const float*)d_in[8];
    const float* conv_w  = (const float*)d_in[9];
    const float* conv_b  = (const float*)d_in[10];
    const float* xprj    = (const float*)d_in[11];
    const float* dtw     = (const float*)d_in[12];
    const float* dtb     = (const float*)d_in[13];
    const float* alog    = (const float*)d_in[14];
    const float* Dv      = (const float*)d_in[15];
    const float* outw    = (const float*)d_in[16];
    const float* lng     = (const float*)d_in[17];
    const float* lnb     = (const float*)d_in[18];
    const float* headw   = (const float*)d_in[19];
    const float* headb   = (const float*)d_in[20];
    float* out = (float*)d_out;

    uint8_t* w = (uint8_t*)d_ws;
    size_t off = 0;
    auto alloc = [&](size_t bytes) -> uint8_t* {
        uint8_t* p = w + off;
        off += (bytes + 255) & ~(size_t)255;
        return p;
    };
    _Float16* ws_state_h = (_Float16*)alloc((size_t)MROWS * SDIM * 2);
    float*    ws_tok_p   = (float*)   alloc((size_t)MROWS * DM * 4);
    _Float16* ws_tok_h   = (_Float16*)alloc((size_t)MROWS * DM * 2);
    _Float16* ws_inprj_h = (_Float16*)alloc((size_t)2 * DIN * DM * 2);
    _Float16* ws_xprj_h  = (_Float16*)alloc((size_t)(DTR + 2 * DST) * DIN * 2);
    _Float16* ws_Ws_h    = (_Float16*)alloc((size_t)DM * SDIM * 2);
    float*    ws_aneg    = (float*)   alloc((size_t)DIN * DST * 4);
    _Float16* ws_xz_h    = (_Float16*)alloc((size_t)MROWS * 2 * DIN * 2);
    _Float16* ws_xconv_h = (_Float16*)alloc((size_t)MROWS * DIN * 2);
    float*    ws_dbc     = (float*)   alloc((size_t)MROWS * (DTR + 2 * DST) * 4);
    float*    ws_dt      = (float*)   alloc((size_t)MROWS * DIN * 4);
    _Float16* ws_yg_h    = (_Float16*)alloc((size_t)MROWS * DIN * 2);
    int*      ws_idx     = (int*)     alloc(BB * 4);

    const int NBC = DTR + 2 * DST;   // 48

    int n;
    n = MROWS * SDIM;
    cvt_f32_to_f16_kernel<<<(n + 255) / 256, 256, 0, stream>>>(state, ws_state_h, n);
    n = DM * SDIM;
    cvt_f32_to_f16_kernel<<<(n + 255) / 256, 256, 0, stream>>>(Ws, ws_Ws_h, n);
    n = 2 * DIN * DM;
    cvt_f32_to_f16_kernel<<<(n + 255) / 256, 256, 0, stream>>>(inprj, ws_inprj_h, n);
    n = NBC * DIN;
    cvt_f32_to_f16_kernel<<<(n + 255) / 256, 256, 0, stream>>>(xprj, ws_xprj_h, n);
    n = DIN * DST;
    aneg_kernel<<<(n + 255) / 256, 256, 0, stream>>>(alog, ws_aneg, n);
    lastidx_kernel<<<BB, 256, 0, stream>>>(mask, ws_idx);

    // embed: state @ Ws^T via WMMA, then rank-1 rtg + biases + pos + mask
    gemm_wmma_f16_kernel<<<dim3(MROWS / 32, DM / 64), 256, 0, stream>>>(
        ws_state_h, ws_Ws_h, ws_tok_p, nullptr, MROWS, DM, SDIM);
    n = MROWS * DM;
    embed_epilogue_kernel<<<(n + 255) / 256, 256, 0, stream>>>(
        ws_tok_p, rtg, mask, Wr, bs, br, pos, ws_tok_h, n);

    // in_proj: TDM-staged weights in LDS + 2x2 WMMA tiles per wave
    gemm_inproj_wmma_tdm_kernel<<<dim3(MROWS / 64, (2 * DIN) / IP_NT), 256, 0, stream>>>(
        ws_tok_h, ws_inprj_h, ws_xz_h, MROWS, 2 * DIN, /*never=*/0);

    // causal depthwise conv + SiLU
    n = MROWS * DIN;
    conv_silu_kernel<<<(n + 255) / 256, 256, 0, stream>>>(
        ws_xz_h, conv_w, conv_b, ws_xconv_h, n);

    // x_proj: (16384 x 512) @ (48 x 512)^T -> dbc (f32), WMMA
    gemm_wmma_f16_kernel<<<dim3(MROWS / 32, 1), 256, 0, stream>>>(
        ws_xconv_h, ws_xprj_h, ws_dbc, nullptr, MROWS, NBC, DIN);

    // dt = softplus(dtr @ dt_proj^T + b)
    n = MROWS * DIN;
    dt_kernel<<<(n + 255) / 256, 256, 0, stream>>>(ws_dbc, dtw, dtb, ws_dt, n);

    // selective scan (fused dA/dBx/D-skip/SiLU gate) -> yg (f16)
    scan_kernel<<<BB * 2, 256, 0, stream>>>(
        ws_dt, ws_xconv_h, ws_xz_h, ws_dbc, ws_aneg, Dv, ws_yg_h);

    // out_proj + LayerNorm + head, only for the 8 selected rows
    head_kernel<<<BB, 256, 0, stream>>>(
        ws_yg_h, outw, lng, lnb, headw, headb, ws_idx, out);

    (void)in_sizes; (void)n_in; (void)out_size; (void)ws_size;
}
